// VisionLanguageActionModel_35854386987598
// MI455X (gfx1250) — matrix operand order repaired and checked by reading, not compiled, not run on hardware
//
#include <hip/hip_runtime.h>
#include <hip/hip_bf16.h>

typedef __attribute__((ext_vector_type(16))) _Float16 v16h;
typedef __attribute__((ext_vector_type(8)))  _Float16 v8h;
typedef __attribute__((ext_vector_type(8)))  float    v8f;

#define B_  2
#define S_  2048
#define D_  1024
#define DS_ 16
#define K_  4
#define L_  6
#define DI_ 2048
#define V_  32000
#define A_  7
#define M_  (B_*S_)   // 4096 rows for every GEMM

// Async DMA global->LDS (CDNA5 GLOBAL_LOAD_ASYNC_TO_LDS_B128, ASYNCcnt-tracked).
// ldsoff: LDS byte offset (low 32 bits of the generic pointer = LDS offset per
// the flat-aperture rules); gaddr: 64-bit global address.
#define ASYNC_LDS_B128(ldsoff, gaddr)                                   \
    asm volatile("global_load_async_to_lds_b128 %0, %1, off"            \
                 :: "v"(ldsoff), "v"(gaddr) : "memory")
#define WAIT_ASYNC0() asm volatile("s_wait_asynccnt 0x0" ::: "memory")

// ---------------------------------------------------------------------------
// WMMA GEMM: C[M,N] = act( A[M,K]_f16 @ (Bt[N,K]_f16)^T + bias + resid )
// Block tile 128x128, 8 waves (4x2), each wave 2x4 accumulators of 16x16.
// Double-buffered LDS filled by async-DMA, overlapped with WMMA compute.
// ---------------------------------------------------------------------------
__global__ __launch_bounds__(256)
void gemm_wmma_f16(const _Float16* __restrict__ A,
                   const _Float16* __restrict__ Bt,
                   const float* __restrict__ bias,
                   const float* __restrict__ resid,
                   float* __restrict__ C,
                   int M, int N, int K, int act)
{
    __shared__ __align__(16) _Float16 As[2][128 * 32];
    __shared__ __align__(16) _Float16 Bs[2][128 * 32];

    const int bm   = blockIdx.y * 128;
    const int bn   = blockIdx.x * 128;
    const int tid  = threadIdx.x;
    const int lane = tid & 31;
    const int wave = tid >> 5;
    const int wm   = (wave >> 1) * 32;   // wave row offset in block tile
    const int wn   = (wave & 1) * 64;    // wave col offset in block tile
    const int frow  = lane & 15;         // row (A) / col (B,C) within 16
    const int khalf = lane >> 4;         // which K-half this lane holds

    // Per-thread staging coordinates: rows r and r+64, 8-half column chunk co.
    const int r  = tid >> 2;
    const int co = (tid & 3) << 3;

    // LDS byte offsets of this thread's four DMA destinations, per buffer.
    unsigned dAs0[2], dAs1[2], dBs0[2], dBs1[2];
    #pragma unroll
    for (int b = 0; b < 2; ++b) {
        dAs0[b] = (unsigned)(size_t)&As[b][r * 32 + co];
        dAs1[b] = (unsigned)(size_t)&As[b][(r + 64) * 32 + co];
        dBs0[b] = (unsigned)(size_t)&Bs[b][r * 32 + co];
        dBs1[b] = (unsigned)(size_t)&Bs[b][(r + 64) * 32 + co];
    }
    const _Float16* gA0 = A  + (size_t)(bm + r)      * K + co;
    const _Float16* gA1 = A  + (size_t)(bm + r + 64) * K + co;
    const _Float16* gB0 = Bt + (size_t)(bn + r)      * K + co;
    const _Float16* gB1 = Bt + (size_t)(bn + r + 64) * K + co;

    auto stage = [&](int buf, int k0) {
        ASYNC_LDS_B128(dAs0[buf], (unsigned long long)(size_t)(gA0 + k0));
        ASYNC_LDS_B128(dAs1[buf], (unsigned long long)(size_t)(gA1 + k0));
        ASYNC_LDS_B128(dBs0[buf], (unsigned long long)(size_t)(gB0 + k0));
        ASYNC_LDS_B128(dBs1[buf], (unsigned long long)(size_t)(gB1 + k0));
    };

    v8f acc[2][4];
    #pragma unroll
    for (int i = 0; i < 2; ++i)
        #pragma unroll
        for (int j = 0; j < 4; ++j)
            #pragma unroll
            for (int e = 0; e < 8; ++e) acc[i][j][e] = 0.0f;

    const int nk = K >> 5;
    stage(0, 0);                 // prologue: DMA tile 0 into buffer 0
    WAIT_ASYNC0();
    __syncthreads();

    for (int kt = 0; kt < nk; ++kt) {
        const int buf = kt & 1;
        if (kt + 1 < nk) {
            stage(buf ^ 1, (kt + 1) << 5);   // DMA next tile under compute
            if (kt + 2 < nk) {               // L2 hint for the tile after
                __builtin_prefetch(gA0 + ((kt + 2) << 5), 0, 1);
                __builtin_prefetch(gB0 + ((kt + 2) << 5), 0, 1);
            }
        }

        // Fragment loads per documented CDNA5 16-bit WMMA layouts.
        // A 16x32: lane half 0 -> K 0..7 & 16..23 ; half 1 -> K 8..15 & 24..31
        v16h af[2];
        #pragma unroll
        for (int mi = 0; mi < 2; ++mi) {
            const _Float16* p = &As[buf][(wm + mi * 16 + frow) * 32 + khalf * 8];
            v8h lo = *(const v8h*)p;
            v8h hi = *(const v8h*)(p + 16);
            #pragma unroll
            for (int t = 0; t < 8; ++t) { af[mi][t] = lo[t]; af[mi][t + 8] = hi[t]; }
        }
        // B 32x16: lane half 0 -> K 0..7 & 16..23 ; half 1 -> K 8..15 & 24..31 (column = frow)
        v16h bf[4];
        #pragma unroll
        for (int ni = 0; ni < 4; ++ni) {
            const _Float16* p = &Bs[buf][(wn + ni * 16 + frow) * 32 + khalf * 8];
            v8h lo = *(const v8h*)p;
            v8h hi = *(const v8h*)(p + 16);
            #pragma unroll
            for (int t = 0; t < 8; ++t) { bf[ni][t] = lo[t]; bf[ni][t + 8] = hi[t]; }
        }

        #pragma unroll
        for (int mi = 0; mi < 2; ++mi)
            #pragma unroll
            for (int ni = 0; ni < 4; ++ni)
                acc[mi][ni] = __builtin_amdgcn_wmma_f32_16x16x32_f16(
                    false, af[mi], false, bf[ni], (short)0, acc[mi][ni],
                    false, false);

        if (kt + 1 < nk) WAIT_ASYNC0();   // next tile fully in LDS
        __syncthreads();
    }

    // Epilogue. C/D layout: VGPR v holds M = v + 8*khalf, N = frow.
    #pragma unroll
    for (int mi = 0; mi < 2; ++mi)
        #pragma unroll
        for (int ni = 0; ni < 4; ++ni)
            #pragma unroll
            for (int v = 0; v < 8; ++v) {
                const int rr  = bm + wm + mi * 16 + khalf * 8 + v;
                const int cix = bn + wn + ni * 16 + frow;
                float val = acc[mi][ni][v];
                if (bias)  val += bias[cix];
                if (resid) val += resid[(size_t)rr * N + cix];
                if (act == 1) val = fmaxf(val, 0.0f);
                C[(size_t)rr * N + cix] = val;
            }
}

// ---------------------------------------------------------------------------
// Helpers / small kernels
// ---------------------------------------------------------------------------
__global__ void k_copy(const float* __restrict__ s, float* __restrict__ d, int n) {
    int i = blockIdx.x * blockDim.x + threadIdx.x;
    if (i < n) d[i] = s[i];
}

__global__ void k_f32_to_f16(const float* __restrict__ s, _Float16* __restrict__ d, int n) {
    int i = blockIdx.x * blockDim.x + threadIdx.x;
    if (i < n) d[i] = (_Float16)s[i];
}

// W (KxN f32, row major) -> Wt (NxK f16) tiled transpose
__global__ void k_transpose_f16(const float* __restrict__ W, _Float16* __restrict__ Wt,
                                int K, int N) {
    __shared__ float tile[32][33];
    const int n0 = blockIdx.x * 32, k0 = blockIdx.y * 32;
    const int tx = threadIdx.x & 31, ty = threadIdx.x >> 5;   // ty: 0..7
    #pragma unroll
    for (int i = 0; i < 4; ++i) {
        int k = ty + i * 8;
        tile[k][tx] = W[(size_t)(k0 + k) * N + n0 + tx];
    }
    __syncthreads();
    #pragma unroll
    for (int i = 0; i < 4; ++i) {
        int n = ty + i * 8;
        Wt[(size_t)(n0 + n) * K + k0 + tx] = (_Float16)tile[tx][n];
    }
}

// Causal depthwise conv (K=4) + bias + SiLU on xp = XZ[:, :DI]
__global__ void k_conv_silu(const float* __restrict__ XZ, const float* __restrict__ cw,
                            const float* __restrict__ cb, float* __restrict__ XC) {
    int idx = blockIdx.x * blockDim.x + threadIdx.x;
    if (idx >= M_ * DI_) return;
    const int m = idx / DI_, c = idx % DI_;
    const int b = m / S_, s = m % S_;
    float acc = cb[c];
    #pragma unroll
    for (int k = 0; k < K_; ++k) {
        int sp = s - (K_ - 1) + k;
        if (sp >= 0)
            acc += XZ[(size_t)(b * S_ + sp) * (2 * DI_) + c] * cw[c * K_ + k];
    }
    XC[idx] = acc / (1.0f + expf(-acc));
}

// U[m,n] = 0.1 * sum_i XC[m,i] * Bm[i,n]
__global__ void k_u(const float* __restrict__ XC, const float* __restrict__ Bm,
                    float* __restrict__ U) {
    int idx = blockIdx.x * blockDim.x + threadIdx.x;
    if (idx >= M_ * DS_) return;
    const int m = idx / DS_, n = idx % DS_;
    float acc = 0.0f;
    for (int i = 0; i < DI_; ++i)
        acc += XC[(size_t)m * DI_ + i] * Bm[i * DS_ + n];
    U[idx] = acc * 0.1f;
}

// 32 independent recurrence chains h = 0.9h + u (unrolled so loads pipeline)
__global__ void k_scan(const float* __restrict__ U, float* __restrict__ HS) {
    const int t = threadIdx.x;          // 0..31
    const int b = t >> 4, n = t & 15;
    float h = 0.0f;
    #pragma unroll 4
    for (int s = 0; s < S_; ++s) {
        h = 0.9f * h + U[(size_t)(b * S_ + s) * DS_ + n];
        HS[(size_t)(b * S_ + s) * DS_ + n] = h;
    }
}

// Y[m,i] = (HS[m,:] @ Cm[:,i]) * silu(z[m,i]),  z = XZ[:, DI:]
__global__ void k_y(const float* __restrict__ HS, const float* __restrict__ Cm,
                    const float* __restrict__ XZ, float* __restrict__ Y) {
    int idx = blockIdx.x * blockDim.x + threadIdx.x;
    if (idx >= M_ * DI_) return;
    const int m = idx / DI_, i = idx % DI_;
    float acc = 0.0f;
    #pragma unroll
    for (int n = 0; n < DS_; ++n)
        acc += HS[(size_t)m * DS_ + n] * Cm[n * DI_ + i];
    const float z = XZ[(size_t)m * (2 * DI_) + DI_ + i];
    Y[idx] = acc * (z / (1.0f + expf(-z)));
}

// LayerNorm over D, emit f16 directly (only consumed by WMMA GEMMs)
__global__ void k_layernorm_f16(const float* __restrict__ X, const float* __restrict__ g,
                                const float* __restrict__ bta, _Float16* __restrict__ out) {
    __shared__ float r1[256], r2[256];
    const int row = blockIdx.x, t = threadIdx.x;
    const float* x = X + (size_t)row * D_;
    float s = 0.0f, s2 = 0.0f;
    for (int i = t; i < D_; i += 256) { float v = x[i]; s += v; s2 += v * v; }
    r1[t] = s; r2[t] = s2; __syncthreads();
    for (int o = 128; o > 0; o >>= 1) {
        if (t < o) { r1[t] += r1[t + o]; r2[t] += r2[t + o]; }
        __syncthreads();
    }
    const float mu  = r1[0] / D_;
    const float var = r2[0] / D_ - mu * mu;
    const float rs  = rsqrtf(var + 1e-5f);
    for (int i = t; i < D_; i += 256)
        out[(size_t)row * D_ + i] = (_Float16)((x[i] - mu) * rs * g[i] + bta[i]);
}

// actions = tanh(H2 @ act_w3 + b3)   (K=256, N=7 — too small for WMMA)
__global__ void k_actions(const float* __restrict__ H2, const float* __restrict__ w3,
                          const float* __restrict__ b3, float* __restrict__ out) {
    int idx = blockIdx.x * blockDim.x + threadIdx.x;
    if (idx >= M_ * A_) return;
    const int m = idx / A_, a = idx % A_;
    float acc = b3[a];
    for (int j = 0; j < D_ / 4; ++j)
        acc += H2[(size_t)m * (D_ / 4) + j] * w3[j * A_ + a];
    out[idx] = tanhf(acc);
}

// ---------------------------------------------------------------------------
extern "C" void kernel_launch(void* const* d_in, const int* in_sizes, int n_in,
                              void* d_out, int out_size, void* d_ws, size_t ws_size,
                              hipStream_t stream) {
    (void)in_sizes; (void)n_in; (void)out_size; (void)ws_size;
    const float* features = (const float*)d_in[0];
    const float* in_proj  = (const float*)d_in[1];
    const float* conv_w   = (const float*)d_in[2];
    const float* conv_b   = (const float*)d_in[3];
    const float* B_ssm    = (const float*)d_in[4];
    const float* C_ssm    = (const float*)d_in[5];
    const float* out_w    = (const float*)d_in[6];
    const float* ln_g     = (const float*)d_in[7];
    const float* ln_b     = (const float*)d_in[8];
    const float* act_w1   = (const float*)d_in[9];
    const float* act_b1   = (const float*)d_in[10];
    const float* act_w2   = (const float*)d_in[11];
    const float* act_b2   = (const float*)d_in[12];
    const float* act_w3   = (const float*)d_in[13];
    const float* act_b3   = (const float*)d_in[14];
    const float* lang_w1  = (const float*)d_in[15];
    const float* lang_b1  = (const float*)d_in[16];
    const float* lang_w2  = (const float*)d_in[17];
    const float* lang_b2  = (const float*)d_in[18];

    char* ws = (char*)d_ws;
    const size_t MB = (size_t)1 << 20;
    float*    X    = (float*)(ws + 0);          // 16 MB  activations (f32)
    float*    XZ   = (float*)(ws + 16 * MB);    // 64 MB  in_proj output
    float*    XC   = (float*)(ws + 80 * MB);    // 32 MB  conv+silu
    float*    Y    = (float*)(ws + 112 * MB);   // 32 MB  scan output * silu(z)
    float*    U    = (float*)(ws + 144 * MB);   // 256 KB
    float*    HS   = (float*)(ws + 145 * MB);   // 256 KB
    _Float16* ABF  = (_Float16*)(ws + 146 * MB);// 16 MB  f16 A staging
    _Float16* XLNB = (_Float16*)(ws + 162 * MB);// 8 MB   f16 layernorm out
    _Float16* WT   = (_Float16*)(ws + 170 * MB);// 66 MB  f16 transposed weights
    float*    H1   = (float*)(ws + 16 * MB);    // reuse XZ region post-layers
    float*    H2   = (float*)(ws + 32 * MB);
    float*    HL   = (float*)(ws + 48 * MB);

    float* actions = (float*)d_out;
    float* logits  = (float*)d_out + (size_t)M_ * A_;

    const dim3 b256(256);
    auto g1 = [](int n) { return dim3((unsigned)((n + 255) / 256)); };

    k_copy<<<g1(M_ * D_), b256, 0, stream>>>(features, X, M_ * D_);

    for (int l = 0; l < L_; ++l) {
        k_f32_to_f16<<<g1(M_ * D_), b256, 0, stream>>>(X, ABF, M_ * D_);
        k_transpose_f16<<<dim3((2 * DI_) / 32, D_ / 32), b256, 0, stream>>>(
            in_proj + (size_t)l * D_ * 2 * DI_, WT, D_, 2 * DI_);
        gemm_wmma_f16<<<dim3((2 * DI_) / 128, M_ / 128), b256, 0, stream>>>(
            ABF, WT, nullptr, nullptr, XZ, M_, 2 * DI_, D_, 0);

        k_conv_silu<<<g1(M_ * DI_), b256, 0, stream>>>(
            XZ, conv_w + (size_t)l * DI_ * K_, conv_b + (size_t)l * DI_, XC);
        k_u<<<g1(M_ * DS_), b256, 0, stream>>>(XC, B_ssm + (size_t)l * DI_ * DS_, U);
        k_scan<<<dim3(1), dim3(32), 0, stream>>>(U, HS);
        k_y<<<g1(M_ * DI_), b256, 0, stream>>>(HS, C_ssm + (size_t)l * DS_ * DI_, XZ, Y);

        k_f32_to_f16<<<g1(M_ * DI_), b256, 0, stream>>>(Y, ABF, M_ * DI_);
        k_transpose_f16<<<dim3(D_ / 32, DI_ / 32), b256, 0, stream>>>(
            out_w + (size_t)l * DI_ * D_, WT, DI_, D_);
        gemm_wmma_f16<<<dim3(D_ / 128, M_ / 128), b256, 0, stream>>>(
            ABF, WT, nullptr, /*resid=*/X, X, M_, D_, DI_, 0);
    }

    k_layernorm_f16<<<dim3(M_), b256, 0, stream>>>(X, ln_g, ln_b, XLNB);

    // action head: relu GEMMs then tiny tanh projection
    k_transpose_f16<<<dim3((D_ / 2) / 32, D_ / 32), b256, 0, stream>>>(act_w1, WT, D_, D_ / 2);
    gemm_wmma_f16<<<dim3((D_ / 2) / 128, M_ / 128), b256, 0, stream>>>(
        XLNB, WT, act_b1, nullptr, H1, M_, D_ / 2, D_, 1);
    k_f32_to_f16<<<g1(M_ * D_ / 2), b256, 0, stream>>>(H1, ABF, M_ * D_ / 2);
    k_transpose_f16<<<dim3((D_ / 4) / 32, (D_ / 2) / 32), b256, 0, stream>>>(act_w2, WT, D_ / 2, D_ / 4);
    gemm_wmma_f16<<<dim3((D_ / 4) / 128, M_ / 128), b256, 0, stream>>>(
        ABF, WT, act_b2, nullptr, H2, M_, D_ / 4, D_ / 2, 1);
    k_actions<<<g1(M_ * A_), b256, 0, stream>>>(H2, act_w3, act_b3, actions);

    // language head: relu(x @ W1 + b1) @ W2 + b2  (the 268 GF GEMM)
    k_transpose_f16<<<dim3(D_ / 32, D_ / 32), b256, 0, stream>>>(lang_w1, WT, D_, D_);
    gemm_wmma_f16<<<dim3(D_ / 128, M_ / 128), b256, 0, stream>>>(
        XLNB, WT, lang_b1, nullptr, HL, M_, D_, D_, 1);
    k_f32_to_f16<<<g1(M_ * D_), b256, 0, stream>>>(HL, ABF, M_ * D_);
    k_transpose_f16<<<dim3(V_ / 32, D_ / 32), b256, 0, stream>>>(lang_w2, WT, D_, V_);
    gemm_wmma_f16<<<dim3(V_ / 128, M_ / 128), b256, 0, stream>>>(
        ABF, WT, lang_b2, nullptr, logits, M_, V_, D_, 0);
}
